// T2FN_1992864825695
// MI455X (gfx1250) — matrix operand order repaired and checked by reading, not compiled, not run on hardware
//
#include <hip/hip_runtime.h>
#include <hip/hip_bf16.h>
#include <math.h>

typedef __attribute__((ext_vector_type(16))) _Float16 v16h;
typedef __attribute__((ext_vector_type(8)))  float    v8f;

#define B_        128
#define T_        64
#define PF_       128
#define NAV       2401    // 49*49 audio-video pairs
#define NCHI      97      // text dim + 1
#define CHIP      112     // chi padded to 7*16
#define NCHUNK    151     // ceil(2401/16)
#define FD_STRIDE 264     // fusion/W1 LDS k-stride in halfs (256+8, 16B-aligned rows)
#define AV_STRIDE 72      // av tile row stride in halfs (16B-aligned, bank-spread)

static __device__ __forceinline__ v16h pack2(uint4 lo, uint4 hi) {
  union { uint4 q[2]; v16h h; } u;
  u.q[0] = lo; u.q[1] = hi;
  return u.h;
}

// A fragment (16 M x 32 K, f16) from a row-major 16-row tile.
// ISA layout: lanes 0-15: M=lane, K=[0,8)+[16,24); lanes 16-31: M=lane-16, K=[8,16)+[24,32).
static __device__ __forceinline__ v16h fragA(const _Float16* base, int stride, int k0) {
  int lane = threadIdx.x & 31;
  const _Float16* r = base + (lane & 15) * stride + k0 + ((lane >> 4) << 3);
  uint4 lo = *(const uint4*)(r);
  uint4 hi = *(const uint4*)(r + 16);
  return pack2(lo, hi);
}

// B fragment (32 K x 16 N, f16) from an N-major (transposed) tile.
// lanes 0-15: N=lane, K=[0,8)+[16,24); lanes 16-31: N=lane-16, K=[8,16)+[24,32).
static __device__ __forceinline__ v16h fragB(const _Float16* base, int stride, int k0) {
  int lane = threadIdx.x & 31;
  const _Float16* r = base + (lane & 15) * stride + k0 + ((lane >> 4) << 3);
  uint4 lo = *(const uint4*)(r);
  uint4 hi = *(const uint4*)(r + 16);
  return pack2(lo, hi);
}

// ---------------------------------------------------------------------------
// Kernel 1: zero h1 accumulator; build transposed padded text tile xt[b][chi][t]
// (chi=0 -> 1.0, chi 1..96 -> text, chi 97..111 -> 0) as f16 for B-fragment loads.
// ---------------------------------------------------------------------------
__global__ void tfn_init(const float* __restrict__ text,
                         float* __restrict__ h1acc,
                         _Float16* __restrict__ xt) {
  int idx = blockIdx.x * 256 + threadIdx.x;        // grid covers 128*112*64
  if (idx < B_ * PF_) h1acc[idx] = 0.f;
  if (idx < B_ * CHIP * T_) {
    int t   = idx & 63;
    int rem = idx >> 6;
    int ch  = rem % CHIP;
    int b   = rem / CHIP;
    float v = 0.f;
    if (ch == 0)        v = 1.f;
    else if (ch < NCHI) v = text[(b * T_ + t) * 96 + (ch - 1)];
    xt[idx] = (_Float16)v;
  }
}

// ---------------------------------------------------------------------------
// Kernel 2: fused fusion-build + fusion @ W1.
// Each workgroup (8 waves) processes 16 av-pairs per chunk iteration:
//   for each chi-subtile of 16:
//     - coop load W1 rows k=(p*97+chi) -> N-major f16 LDS (zero padded)
//     - per wave (owns 16 batches): build av[16x64] f16 in LDS, 2 WMMAs ->
//       fusion tile 16x16 f32 -> f16 scatter into wave-private fusion rows
//     - per wave: 64 WMMAs C[16x128] += Fus[16x256] @ W1t[256x128]
// C accumulates in registers across all chunks; epilogue = f32 global atomics.
// ---------------------------------------------------------------------------
__global__ void __launch_bounds__(256) tfn_main(
    const float* __restrict__ audio, const float* __restrict__ video,
    const float* __restrict__ W1, const _Float16* __restrict__ xt,
    float* __restrict__ h1acc) {
  __shared__ _Float16 sW1t[128 * FD_STRIDE];     // 67.6 KB, [n][kk]
  __shared__ _Float16 sFus[128 * FD_STRIDE];     // 67.6 KB, [b][kk]
  __shared__ _Float16 sAv[8 * 16 * AV_STRIDE];   // 18.4 KB, per-wave [m][t]

  const int tid  = threadIdx.x;
  const int lane = tid & 31;
  const int wid  = tid >> 5;

  v8f acc[8] = {};   // C stripe: rows = wid*16..+15 (batches), cols = 128 PF

  for (int chunk = blockIdx.x; chunk < NCHUNK; chunk += gridDim.x) {
    const int p0 = chunk * 16;
    for (int s = 0; s < 7; ++s) {
      __syncthreads();   // protect W1t against previous sub-iteration's reads
      // ---- cooperative W1 sub-tile load: 256 k-rows x 128 cols -> [n][kk] f16
      #pragma unroll 4
      for (int it = 0; it < 128; ++it) {
        int e  = it * 256 + tid;
        int n  = e & 127;
        int kk = e >> 7;                 // 0..255 = m*16 + cl
        int m  = kk >> 4, cl = kk & 15;
        int p  = p0 + m, ch = s * 16 + cl;
        float v = 0.f;
        if (p < NAV && ch < NCHI) v = W1[(p * NCHI + ch) * PF_ + n];
        sW1t[n * FD_STRIDE + kk] = (_Float16)v;
      }
      __syncthreads();

      // ---- phase 1: per-wave fusion build (WMMA), 16 batches each
      _Float16* avt = sAv  + wid * 16 * AV_STRIDE;
      _Float16* fus = sFus + wid * 16 * FD_STRIDE;
      for (int i = 0; i < 16; ++i) {
        const int b = wid * 16 + i;
        #pragma unroll
        for (int q = 0; q < 32; ++q) {   // av[m][t] = a(alpha)*v(nu), with 1-augment
          int m = q >> 1;
          int t = ((q & 1) << 5) | lane;
          int p = p0 + m;
          float v = 0.f;
          if (p < NAV) {
            int al = p / 49;
            int vl = p - al * 49;
            float fa = al ? audio[(b * T_ + t) * 48 + (al - 1)] : 1.f;
            float fv = vl ? video[(b * T_ + t) * 48 + (vl - 1)] : 1.f;
            v = fa * fv;
          }
          avt[m * AV_STRIDE + t] = (_Float16)v;
        }
        asm volatile("s_wait_dscnt 0" ::: "memory");
        v16h a0 = fragA(avt, AV_STRIDE, 0);
        v16h a1 = fragA(avt, AV_STRIDE, 32);
        const _Float16* xrow = xt + (b * CHIP + s * 16) * T_;  // [chi][t], stride 64
        v16h x0 = fragB(xrow, T_, 0);
        v16h x1 = fragB(xrow, T_, 32);
        v8f d = {};
        d = __builtin_amdgcn_wmma_f32_16x16x32_f16(false, a0, false, x0, (short)0, d, false, false);
        d = __builtin_amdgcn_wmma_f32_16x16x32_f16(false, a1, false, x1, (short)0, d, false, false);
        // scatter D(m=av-local, n=chi-local) -> fusion row b at kk = m*16+n
        int nn = lane & 15, mb = (lane >> 4) << 3;
        #pragma unroll
        for (int j = 0; j < 8; ++j)
          fus[i * FD_STRIDE + (mb + j) * 16 + nn] = (_Float16)d[j];
      }
      asm volatile("s_wait_dscnt 0" ::: "memory");

      // ---- phase 2: C[16x128] += Fus[16x256] @ W1t[256x128]
      v16h af[8];
      #pragma unroll
      for (int ks = 0; ks < 8; ++ks) af[ks] = fragA(fus, FD_STRIDE, ks * 32);
      #pragma unroll
      for (int nt = 0; nt < 8; ++nt) {
        #pragma unroll
        for (int ks = 0; ks < 8; ++ks) {
          v16h bf = fragB(sW1t + (nt * 16) * FD_STRIDE, FD_STRIDE, ks * 32);
          acc[nt] = __builtin_amdgcn_wmma_f32_16x16x32_f16(false, af[ks], false, bf,
                                                           (short)0, acc[nt], false, false);
        }
      }
    }
  }

  // ---- epilogue: accumulate partial C into global h1 accumulator
  int nn = lane & 15, mb = (lane >> 4) << 3;
  #pragma unroll
  for (int nt = 0; nt < 8; ++nt) {
    #pragma unroll
    for (int j = 0; j < 8; ++j)
      atomicAdd(&h1acc[(wid * 16 + mb + j) * PF_ + nt * 16 + nn], acc[nt][j]);
  }
}

// ---------------------------------------------------------------------------
// Kernel 3: MLP tail. h1=relu(h1acc+b1); h2=relu(h1@W2+b2) via WMMA;
// out = sigmoid(h2@W3+b3)*6-3.
// ---------------------------------------------------------------------------
__global__ void __launch_bounds__(256) tfn_tail(
    const float* __restrict__ h1acc, const float* __restrict__ b1,
    const float* __restrict__ W2, const float* __restrict__ b2,
    const float* __restrict__ W3, const float* __restrict__ b3,
    float* __restrict__ out) {
  __shared__ _Float16 sH1[128 * 136];   // [b][k]
  __shared__ _Float16 sW2[128 * 136];   // [n][k] (transposed)
  __shared__ _Float16 sH2[128 * 136];   // [b][n]
  const int tid = threadIdx.x, lane = tid & 31, wid = tid >> 5;

  for (int it = 0; it < 64; ++it) {
    int e = it * 256 + tid;
    int r = e >> 7, c = e & 127;
    sH1[r * 136 + c] = (_Float16)fmaxf(h1acc[e] + b1[c], 0.f);
    sW2[c * 136 + r] = (_Float16)W2[e];            // W2[k=r][n=c] -> [n][k]
  }
  __syncthreads();

  v16h af[4];
  #pragma unroll
  for (int ks = 0; ks < 4; ++ks) af[ks] = fragA(sH1 + wid * 16 * 136, 136, ks * 32);
  int nn = lane & 15, mb = (lane >> 4) << 3;
  #pragma unroll
  for (int nt = 0; nt < 8; ++nt) {
    v8f d = {};
    #pragma unroll
    for (int ks = 0; ks < 4; ++ks) {
      v16h bf = fragB(sW2 + (nt * 16) * 136, 136, ks * 32);
      d = __builtin_amdgcn_wmma_f32_16x16x32_f16(false, af[ks], false, bf, (short)0, d, false, false);
    }
    int n = nt * 16 + nn;
    #pragma unroll
    for (int j = 0; j < 8; ++j)
      sH2[(wid * 16 + mb + j) * 136 + n] = (_Float16)fmaxf(d[j] + b2[n], 0.f);
  }
  __syncthreads();

  if (tid < 128) {
    float s = b3[0];
    for (int p = 0; p < 128; ++p) s += (float)sH2[tid * 136 + p] * W3[p];
    out[tid] = 6.f / (1.f + expf(-s)) - 3.f;
  }
}

extern "C" void kernel_launch(void* const* d_in, const int* in_sizes, int n_in,
                              void* d_out, int out_size, void* d_ws, size_t ws_size,
                              hipStream_t stream) {
  const float* audio = (const float*)d_in[0];
  const float* video = (const float*)d_in[1];
  const float* text  = (const float*)d_in[2];
  const float* W1    = (const float*)d_in[3];
  const float* b1    = (const float*)d_in[4];
  const float* W2    = (const float*)d_in[5];
  const float* b2    = (const float*)d_in[6];
  const float* W3    = (const float*)d_in[7];
  const float* b3    = (const float*)d_in[8];
  float* outp  = (float*)d_out;

  float*    h1acc = (float*)d_ws;                                     // 64 KB
  _Float16* xt    = (_Float16*)((char*)d_ws + B_ * PF_ * sizeof(float)); // 1.75 MB

  tfn_init<<<(B_ * CHIP * T_) / 256, 256, 0, stream>>>(text, h1acc, xt);
  tfn_main<<<NCHUNK, 256, 0, stream>>>(audio, video, W1, xt, h1acc);
  tfn_tail<<<1, 256, 0, stream>>>(h1acc, b1, W2, b2, W3, b3, outp);
}
